// MultiheadSelfAttention_48533130445073
// MI455X (gfx1250) — compile-verified, matrix operand order checked by reading, not run on hardware
//
#include <hip/hip_runtime.h>
#include <hip/hip_bf16.h>
#include <math.h>

// Problem constants (from reference)
#define B_   2
#define T_   2048
#define E_   2048
#define H_   16
#define HKV_ 4
#define DH_  128
#define M_   (B_ * T_)        // 4096
#define NQKV_ 3072            // H*DH + 2*HKV*DH = 2048 + 512 + 512

typedef __bf16 bf16;
typedef __attribute__((ext_vector_type(16))) __bf16 v16bf;
typedef __attribute__((ext_vector_type(8)))  float  v8f;
typedef __attribute__((ext_vector_type(4)))  int    v4i;

union Frag { v16bf v; uint4 u[2]; };

#ifndef __has_builtin
#define __has_builtin(x) 0
#endif
#if __has_builtin(__builtin_amdgcn_global_load_async_to_lds_b128)
#define ASYNC_LDS 1
#else
#define ASYNC_LDS 0
#endif

typedef __attribute__((address_space(1))) v4i gv4i;  // global (prints as __device__)
typedef __attribute__((address_space(3))) v4i lv4i;  // LDS    (prints as __shared__)

#if ASYNC_LDS
__device__ __forceinline__ void async_copy_b128(const bf16* src, bf16* dst_lds) {
  __builtin_amdgcn_global_load_async_to_lds_b128(
      (gv4i*)const_cast<bf16*>(src), (lv4i*)dst_lds, 0, 0);
}
#endif

// ---------------------------------------------------------------------------
// Cast / pack kernels: fp32 -> bf16, weights transposed so WMMA B-operand
// fragment loads are contiguous b128 loads per lane.
// ---------------------------------------------------------------------------
__global__ __launch_bounds__(256) void cast_x_kernel(const float* __restrict__ X,
                                                     bf16* __restrict__ Xb, size_t n) {
  size_t i = (size_t)blockIdx.x * blockDim.x + threadIdx.x;
  if (i < n) Xb[i] = (bf16)X[i];
}

// BT[n][k] = W(k, n) for concatenated [Wq | Wk | Wv], N = 3072, K = 2048
__global__ __launch_bounds__(256) void pack_wqkv_kernel(const float* __restrict__ Wq,
                                                        const float* __restrict__ Wk,
                                                        const float* __restrict__ Wv,
                                                        bf16* __restrict__ BT) {
  size_t i = (size_t)blockIdx.x * blockDim.x + threadIdx.x;
  if (i >= (size_t)NQKV_ * E_) return;
  int n = (int)(i / E_);
  int k = (int)(i % E_);
  float v;
  if (n < 2048)      v = Wq[(size_t)k * 2048 + n];
  else if (n < 2560) v = Wk[(size_t)k * 512 + (n - 2048)];
  else               v = Wv[(size_t)k * 512 + (n - 2560)];
  BT[i] = (bf16)v;
}

__global__ __launch_bounds__(256) void pack_wo_kernel(const float* __restrict__ Wo,
                                                      bf16* __restrict__ WoT) {
  size_t i = (size_t)blockIdx.x * blockDim.x + threadIdx.x;
  if (i >= (size_t)E_ * E_) return;
  int n = (int)(i / E_);
  int k = (int)(i % E_);
  WoT[i] = (bf16)Wo[(size_t)k * E_ + n];
}

// ---------------------------------------------------------------------------
// bf16 WMMA GEMM:  C[M,N] = A[M,K] * BT[N,K]^T + bias(n)
// Block = 256 threads = 8 waves; wave w owns rows [(by*8+w)*16, +16),
// cols [bx*64, +64) as 4 WMMA n-tiles. K stepped by 32.
//
// The 64x32 B tile is shared by all 8 waves -> staged in LDS, double-buffered
// with GLOBAL_LOAD_ASYNC_TO_LDS_B128 (one b128 per thread per tile), so tile
// kt+1 streams in while tile kt feeds WMMAs from ds_load_b128. A fragments are
// per-wave private and register-prefetched one iteration ahead.
// ---------------------------------------------------------------------------
__global__ __launch_bounds__(256)
void gemm_bf16_kernel(const bf16* __restrict__ A, const bf16* __restrict__ BT,
                      const float* __restrict__ b0, const float* __restrict__ b1,
                      const float* __restrict__ b2, int n1, int n2,
                      float* __restrict__ C, int M, int N, int K) {
  __shared__ __align__(16) bf16 Bs[2][64 * 32];
  const int tid  = threadIdx.x;
  const int lane = tid & 31;
  const int wave = tid >> 5;
  const int g    = lane >> 4;
  const int rl   = lane & 15;
  const int m0   = (blockIdx.y * 8 + wave) * 16;
  const int n0   = blockIdx.x * 64;

  // Cooperative B-tile staging: thread tid stages 8 bf16 (one b128).
  const int brow = tid >> 2;          // 0..63  (local n)
  const int bchk = (tid & 3) * 8;     // 0,8,16,24 (local k)
  const bf16* bsrc = BT + (size_t)(n0 + brow) * K + bchk;

  v8f acc[4] = {};
  const bf16* arow = A + (size_t)(m0 + rl) * K;
  const int KT = K >> 5;

  Frag a_cur, a_nxt;
  a_cur.u[0] = *(const uint4*)(arow + g * 8);
  a_cur.u[1] = *(const uint4*)(arow + 16 + g * 8);

#if ASYNC_LDS
  async_copy_b128(bsrc, &Bs[0][brow * 32 + bchk]);
  for (int kt = 0; kt < KT; ++kt) {
    const int nb = kt & 1;
    asm volatile("s_wait_asynccnt 0x0" ::: "memory");
    __syncthreads();
    if (kt + 1 < KT) {
      async_copy_b128(bsrc + (kt + 1) * 32, &Bs[nb ^ 1][brow * 32 + bchk]);
      a_nxt.u[0] = *(const uint4*)(arow + (kt + 1) * 32 + g * 8);
      a_nxt.u[1] = *(const uint4*)(arow + (kt + 1) * 32 + 16 + g * 8);
    }
    const bf16* bt = Bs[nb];
#pragma unroll
    for (int nt = 0; nt < 4; ++nt) {
      Frag b;
      b.u[0] = *(const uint4*)(bt + (nt * 16 + rl) * 32 + g * 8);
      b.u[1] = *(const uint4*)(bt + (nt * 16 + rl) * 32 + 16 + g * 8);
      acc[nt] = __builtin_amdgcn_wmma_f32_16x16x32_bf16(
          false, a_cur.v, false, b.v, (short)0, acc[nt], false, false);
    }
    a_cur = a_nxt;
  }
#else
  for (int kt = 0; kt < KT; ++kt) {
    __syncthreads();
    *(uint4*)&Bs[0][brow * 32 + bchk] = *(const uint4*)(bsrc + kt * 32);
    __syncthreads();
    if (kt + 1 < KT) {
      a_nxt.u[0] = *(const uint4*)(arow + (kt + 1) * 32 + g * 8);
      a_nxt.u[1] = *(const uint4*)(arow + (kt + 1) * 32 + 16 + g * 8);
    }
    const bf16* bt = Bs[0];
#pragma unroll
    for (int nt = 0; nt < 4; ++nt) {
      Frag b;
      b.u[0] = *(const uint4*)(bt + (nt * 16 + rl) * 32 + g * 8);
      b.u[1] = *(const uint4*)(bt + (nt * 16 + rl) * 32 + 16 + g * 8);
      acc[nt] = __builtin_amdgcn_wmma_f32_16x16x32_bf16(
          false, a_cur.v, false, b.v, (short)0, acc[nt], false, false);
    }
    a_cur = a_nxt;
  }
#endif

#pragma unroll
  for (int nt = 0; nt < 4; ++nt) {
    int n = n0 + nt * 16 + rl;
    float bias = (n < n1) ? b0[n] : (n < n2) ? b1[n - n1] : b2[n - n2];
#pragma unroll
    for (int r = 0; r < 8; ++r) {
      int m = m0 + r + 8 * g;               // C layout: VGPR r -> row r / r+8
      C[(size_t)m * N + n] = acc[nt][r] + bias;
    }
  }
}

// ---------------------------------------------------------------------------
// QKV epilogue: RoPE on Q,K (first 64 dims, 32 pairs); write bf16:
//   Q  -> [B, H,   T, DH]   (A-operand friendly, contiguous over d)
//   K  -> [B, HKV, T, DH]   (B-operand of S = Q K^T: col = key, contiguous d)
//   V  -> [B, HKV, DH, T]   (transposed: B-operand of P V: col = d, contiguous keys)
// ---------------------------------------------------------------------------
__device__ __forceinline__ void rope2(float& x, float& y, int pos, int j) {
  // theta_j = 10000^(-j/32) = exp(-j/32 * ln 10000)
  float th = __expf(-(float)j * (0.03125f * 9.2103403719761836f));
  float ang = (float)pos * th;
  float sn, cs;
  __sincosf(ang, &sn, &cs);
  float rx = x * cs - y * sn;
  float ry = x * sn + y * cs;
  x = rx; y = ry;
}

__global__ __launch_bounds__(256)
void qkv_post_kernel(const float* __restrict__ QKV, const int* __restrict__ start_pos,
                     bf16* __restrict__ Qb, bf16* __restrict__ Kb, bf16* __restrict__ Vt) {
  size_t idx = (size_t)blockIdx.x * blockDim.x + threadIdx.x;
  const size_t total = (size_t)M_ * (NQKV_ / 2);
  if (idx >= total) return;
  int m  = (int)(idx / (NQKV_ / 2));
  int n  = (int)(idx % (NQKV_ / 2)) * 2;
  int bb = m >> 11;          // / T_
  int t  = m & (T_ - 1);
  int sp = start_pos[0];
  float x = QKV[(size_t)m * NQKV_ + n];
  float y = QKV[(size_t)m * NQKV_ + n + 1];
  if (n < 2048) {            // Q
    int h = n >> 7, d = n & 127;
    if (d < 64) rope2(x, y, sp + t, d >> 1);
    bf16* q = Qb + (((size_t)bb * H_ + h) * T_ + t) * DH_ + d;
    q[0] = (bf16)x; q[1] = (bf16)y;
  } else if (n < 2560) {     // K
    int nn = n - 2048, hk = nn >> 7, d = nn & 127;
    if (d < 64) rope2(x, y, sp + t, d >> 1);
    bf16* kk = Kb + (((size_t)bb * HKV_ + hk) * T_ + t) * DH_ + d;
    kk[0] = (bf16)x; kk[1] = (bf16)y;
  } else {                   // V (transposed store)
    int nn = n - 2560, hk = nn >> 7, d = nn & 127;
    bf16* v0 = Vt + (((size_t)bb * HKV_ + hk) * DH_ + d) * T_ + t;
    v0[0]  = (bf16)x;
    v0[T_] = (bf16)y;        // d+1 row
  }
}

// ---------------------------------------------------------------------------
// Causal GQA flash attention. One wave per 16-query tile; 32 keys per step:
// two 16x16 S tiles (4 WMMAs each over DH=128) -> online softmax (shfl_xor
// reductions over the 16-lane groups of the C layout) -> P (16x32 bf16) via
// per-wave LDS tile -> 8 PV WMMAs accumulating O (16x128 f32). V fragments
// are loaded right after the S WMMAs so they fly under the softmax VALU work.
// ---------------------------------------------------------------------------
__global__ __launch_bounds__(128)
void attn_kernel(const bf16* __restrict__ Q, const bf16* __restrict__ Kb,
                 const bf16* __restrict__ Vt, bf16* __restrict__ Oa) {
  __shared__ __align__(16) bf16 Pl[4][16 * 32];
  const int lane = threadIdx.x & 31;
  const int wave = threadIdx.x >> 5;
  const int g    = lane >> 4;
  const int rl   = lane & 15;
  const int q0   = (blockIdx.x * 4 + wave) * 16;
  const int h    = blockIdx.y;
  const int b    = blockIdx.z;
  const int hkv  = h >> 2;                       // group = H/HKV = 4
  const float scale = 0.08838834764831845f;      // 1/sqrt(128)

  // Q fragments (A operand), 4 x (16x32) covering DH=128
  Frag qa[4];
  {
    const bf16* qr = Q + (((size_t)b * H_ + h) * T_ + q0 + rl) * DH_;
#pragma unroll
    for (int f = 0; f < 4; ++f) {
      qa[f].u[0] = *(const uint4*)(qr + f * 32 + g * 8);
      qa[f].u[1] = *(const uint4*)(qr + f * 32 + 16 + g * 8);
    }
  }

  v8f o[8] = {};
  float mrow[8], lrow[8];
#pragma unroll
  for (int r = 0; r < 8; ++r) { mrow[r] = -1e30f; lrow[r] = 0.0f; }

  const bf16* kbase = Kb + (size_t)(b * HKV_ + hkv) * T_ * DH_;
  const bf16* vbase = Vt + (size_t)(b * HKV_ + hkv) * DH_ * T_;
  bf16* pl = Pl[wave];

  for (int k0 = 0; k0 <= q0 + 15; k0 += 32) {
    // ---- S = Q K^T (two 16x16 tiles) ----
    v8f s[2];
#pragma unroll
    for (int c = 0; c < 2; ++c) {
      const bf16* kr = kbase + (size_t)(k0 + c * 16 + rl) * DH_;
      v8f sc = {};
#pragma unroll
      for (int f = 0; f < 4; ++f) {
        Frag kb;
        kb.u[0] = *(const uint4*)(kr + f * 32 + g * 8);
        kb.u[1] = *(const uint4*)(kr + f * 32 + 16 + g * 8);
        sc = __builtin_amdgcn_wmma_f32_16x16x32_bf16(
            false, qa[f].v, false, kb.v, (short)0, sc, false, false);
      }
      s[c] = sc;
    }
    // ---- V fragments issued early: overlap softmax + LDS round-trip ----
    Frag vb[8];
#pragma unroll
    for (int dt = 0; dt < 8; ++dt) {
      const bf16* vr = vbase + (size_t)(dt * 16 + rl) * T_ + k0;
      vb[dt].u[0] = *(const uint4*)(vr + g * 8);
      vb[dt].u[1] = *(const uint4*)(vr + 16 + g * 8);
    }
    // ---- scale + causal mask ----
#pragma unroll
    for (int c = 0; c < 2; ++c) {
      int key = k0 + c * 16 + rl;
#pragma unroll
      for (int r = 0; r < 8; ++r) {
        int qrow = q0 + r + 8 * g;
        float sv = s[c][r] * scale;
        s[c][r] = (key <= qrow) ? sv : -1e30f;
      }
    }
    // ---- online softmax (row = r + 8*g, cols across the 16-lane group) ----
    float pscale[8];
#pragma unroll
    for (int r = 0; r < 8; ++r) {
      float rmax = fmaxf(s[0][r], s[1][r]);
#pragma unroll
      for (int off = 8; off > 0; off >>= 1)
        rmax = fmaxf(rmax, __shfl_xor(rmax, off, 32));
      float mnew = fmaxf(mrow[r], rmax);
      float corr = __expf(mrow[r] - mnew);
      float p0 = __expf(s[0][r] - mnew);
      float p1 = __expf(s[1][r] - mnew);
      s[0][r] = p0; s[1][r] = p1;
      float rsum = p0 + p1;
#pragma unroll
      for (int off = 8; off > 0; off >>= 1)
        rsum += __shfl_xor(rsum, off, 32);
      lrow[r] = lrow[r] * corr + rsum;
      mrow[r] = mnew;
      pscale[r] = corr;
    }
#pragma unroll
    for (int dt = 0; dt < 8; ++dt)
#pragma unroll
      for (int r = 0; r < 8; ++r) o[dt][r] *= pscale[r];

    // ---- P (C layout, f32) -> LDS bf16 tile -> A-operand fragment ----
#pragma unroll
    for (int c = 0; c < 2; ++c)
#pragma unroll
      for (int r = 0; r < 8; ++r)
        pl[(r + 8 * g) * 32 + c * 16 + rl] = (bf16)s[c][r];
    asm volatile("s_wait_dscnt 0x0" ::: "memory");   // cross-lane LDS RAW
    Frag pf;
    pf.u[0] = *(const uint4*)(pl + rl * 32 + g * 8);
    pf.u[1] = *(const uint4*)(pl + rl * 32 + 16 + g * 8);

    // ---- O += P V ----
#pragma unroll
    for (int dt = 0; dt < 8; ++dt) {
      o[dt] = __builtin_amdgcn_wmma_f32_16x16x32_bf16(
          false, pf.v, false, vb[dt].v, (short)0, o[dt], false, false);
    }
  }

  // ---- normalize and store attention output as bf16 [B, T, E] ----
#pragma unroll
  for (int r = 0; r < 8; ++r) {
    int qrow = q0 + r + 8 * g;
    float inv = 1.0f / lrow[r];
    bf16* orow = Oa + ((size_t)b * T_ + qrow) * E_ + h * DH_;
#pragma unroll
    for (int dt = 0; dt < 8; ++dt)
      orow[dt * 16 + rl] = (bf16)(o[dt][r] * inv);
  }
}

// ---------------------------------------------------------------------------
// Launch. Workspace layout (~124 MB; whole working set is L2-resident on
// MI455X's 192 MB L2, so the multi-pass structure costs little bandwidth):
// ---------------------------------------------------------------------------
extern "C" void kernel_launch(void* const* d_in, const int* in_sizes, int n_in,
                              void* d_out, int out_size, void* d_ws, size_t ws_size,
                              hipStream_t stream) {
  const float* x   = (const float*)d_in[0];
  const float* Wq  = (const float*)d_in[1];
  const float* bq  = (const float*)d_in[2];
  const float* Wk  = (const float*)d_in[3];
  const float* bk  = (const float*)d_in[4];
  const float* Wv  = (const float*)d_in[5];
  const float* bv  = (const float*)d_in[6];
  const float* Wo  = (const float*)d_in[7];
  const float* bo  = (const float*)d_in[8];
  const int*   sp  = (const int*)d_in[9];
  float* out = (float*)d_out;

  bf16*  xb    = (bf16*)d_ws;                                  // M*E bf16
  bf16*  wqkvT = xb    + (size_t)M_ * E_;                      // NQKV*E bf16
  bf16*  woT   = wqkvT + (size_t)NQKV_ * E_;                   // E*E bf16
  float* qkvf  = (float*)(woT + (size_t)E_ * E_);              // M*NQKV f32
  bf16*  Qb    = (bf16*)(qkvf + (size_t)M_ * NQKV_);           // B*H*T*DH
  bf16*  Kb    = Qb + (size_t)B_ * H_   * T_ * DH_;            // B*HKV*T*DH
  bf16*  Vt    = Kb + (size_t)B_ * HKV_ * T_ * DH_;            // B*HKV*DH*T
  bf16*  Oa    = Vt + (size_t)B_ * HKV_ * T_ * DH_;            // M*E bf16

  // 1) casts / weight packing (transpose for contiguous B-operand loads)
  {
    size_t n = (size_t)M_ * E_;
    cast_x_kernel<<<dim3((unsigned)((n + 255) / 256)), dim3(256), 0, stream>>>(x, xb, n);
  }
  {
    size_t n = (size_t)NQKV_ * E_;
    pack_wqkv_kernel<<<dim3((unsigned)((n + 255) / 256)), dim3(256), 0, stream>>>(Wq, Wk, Wv, wqkvT);
  }
  {
    size_t n = (size_t)E_ * E_;
    pack_wo_kernel<<<dim3((unsigned)((n + 255) / 256)), dim3(256), 0, stream>>>(Wo, woT);
  }

  // 2) fused QKV projection: [4096 x 2048] @ [2048 x 3072] + bias
  gemm_bf16_kernel<<<dim3(NQKV_ / 64, M_ / 128), dim3(256), 0, stream>>>(
      xb, wqkvT, bq, bk, bv, /*n1=*/2048, /*n2=*/2560, qkvf, M_, NQKV_, E_);

  // 3) RoPE + layout conversion to bf16 Q/K/V
  {
    size_t n = (size_t)M_ * (NQKV_ / 2);
    qkv_post_kernel<<<dim3((unsigned)((n + 255) / 256)), dim3(256), 0, stream>>>(qkvf, sp, Qb, Kb, Vt);
  }

  // 4) causal GQA flash attention
  attn_kernel<<<dim3(T_ / 64, H_, B_), dim3(128), 0, stream>>>(Qb, Kb, Vt, Oa);

  // 5) output projection: [4096 x 2048] @ [2048 x 2048] + bo -> f32 out
  gemm_bf16_kernel<<<dim3(E_ / 64, M_ / 128), dim3(256), 0, stream>>>(
      Oa, woT, bo, bo, bo, /*n1=*/E_, /*n2=*/E_, out, M_, E_, E_);
}
